// ViT_62294205661670
// MI455X (gfx1250) — matrix operand order repaired and checked by reading, not compiled, not run on hardware
//
#include <hip/hip_runtime.h>
#include <math.h>

typedef _Float16 h16;
typedef __attribute__((ext_vector_type(8)))  _Float16 v8h;
typedef __attribute__((ext_vector_type(16))) _Float16 v16h;
typedef __attribute__((ext_vector_type(8)))  float    v8f;
typedef __attribute__((ext_vector_type(4)))  int      v4i;

#define DEV static __device__ __forceinline__

#ifndef __has_builtin
#define __has_builtin(x) 0
#endif
#if __has_builtin(__builtin_amdgcn_global_load_async_to_lds_b128) && \
    __has_builtin(__builtin_amdgcn_s_wait_asynccnt)
#define HAS_ASYNC 1
#else
#define HAS_ASYNC 0
#endif

#if HAS_ASYNC
typedef __attribute__((address_space(1))) v4i as1_v4i;
typedef __attribute__((address_space(3))) v4i as3_v4i;
DEV void async_cp16(const h16* g, h16* l) {
  __builtin_amdgcn_global_load_async_to_lds_b128((as1_v4i*)g, (as3_v4i*)l, 0, 0);
}
#endif

DEV void wait_async_all() {
#if HAS_ASYNC
  __builtin_amdgcn_s_wait_asynccnt(0);
#endif
}

// ---------------------------------------------------------------------------
// WMMA helpers (V_WMMA_F32_16X16X32_F16, wave32)
// C/D layout: lane 0-15 -> col=lane, rows 0-7 in vgpr 0-7; lane 16-31 -> rows 8-15.
// A (16x32 f16): lane holds row (lane&15); K chunks at [8*hi .. +8) and [16+8*hi .. +8).
// B (32x16 f16): lane holds col (lane&15); K chunks at [8*hi .. +8) and [16+8*hi .. +8).
// ---------------------------------------------------------------------------
DEV v8f wmma_f16(v16h a, v16h b, v8f c) {
  return __builtin_amdgcn_wmma_f32_16x16x32_f16(false, a, false, b, (short)0, c,
                                                false, false);
}

DEV v16h ldfragA(const h16* row, int hi) {
  v8h a = *(const v8h*)(row + hi * 8);
  v8h b = *(const v8h*)(row + 16 + hi * 8);
  v16h r;
#pragma unroll
  for (int e = 0; e < 8; ++e) { r[e] = a[e]; r[e + 8] = b[e]; }
  return r;
}

DEV v16h ldfragB(const h16* row, int hi) {
  v8h a = *(const v8h*)(row + hi * 8);
  v8h b = *(const v8h*)(row + 16 + hi * 8);
  v16h r;
#pragma unroll
  for (int e = 0; e < 8; ++e) { r[e] = a[e]; r[e + 8] = b[e]; }
  return r;
}

// ---------------------------------------------------------------------------
// Generic GEMM: D[M][N] (f32) = A[M][K] (f16, row-major) x Bt[N][K]^T (f16) + bias[N]
// 256 threads (8 waves), tile 128x128, BK=32, double-buffered LDS, async fills
// (gfx1250 GLOBAL_LOAD_ASYNC_TO_LDS_B128) pipelined one tile ahead.
// Wave -> 64x32 (4x2 WMMA tiles). N multiple of 128, K multiple of 32, M arbitrary
// (rows clamped; garbage rows feed only unstored C rows).
// ---------------------------------------------------------------------------
__global__ __launch_bounds__(256) void k_gemm(const h16* __restrict__ A,
                                              const h16* __restrict__ Bt,
                                              const float* __restrict__ bias,
                                              float* __restrict__ D,
                                              int M, int N, int K) {
  __shared__ __align__(16) h16 As[2 * 128 * 32];
  __shared__ __align__(16) h16 Bs[2 * 128 * 32];
  const int n0 = blockIdx.x * 128;
  const int m0 = blockIdx.y * 128;
  const int wave = threadIdx.x >> 5;
  const int lane = threadIdx.x & 31;
  const int hi = lane >> 4;
  const int l15 = lane & 15;
  const int wm = wave >> 2;   // 0..1 : 64-row band
  const int wn = wave & 3;    // 0..3 : 32-col band

  auto stage = [&](int buf, int k0) {
#pragma unroll
    for (int i = 0; i < 2; ++i) {
      int idx = threadIdx.x + i * 256;        // 512 chunks of 8 f16 per matrix
      int r = idx >> 2;
      int c = (idx & 3) * 8;
      int gm = m0 + r;
      if (gm >= M) gm = M - 1;                // harmless clamp (C rows guarded)
      int gn = n0 + r;
      const h16* ga = A + (size_t)gm * K + k0 + c;
      const h16* gb = Bt + (size_t)gn * K + k0 + c;
      h16* la = As + buf * (128 * 32) + r * 32 + c;
      h16* lb = Bs + buf * (128 * 32) + r * 32 + c;
#if HAS_ASYNC
      async_cp16(ga, la);
      async_cp16(gb, lb);
#else
      *(v8h*)la = *(const v8h*)ga;
      *(v8h*)lb = *(const v8h*)gb;
#endif
    }
  };

  v8f acc[4][2];
#pragma unroll
  for (int mt = 0; mt < 4; ++mt)
#pragma unroll
    for (int nt = 0; nt < 2; ++nt)
#pragma unroll
      for (int r = 0; r < 8; ++r) acc[mt][nt][r] = 0.f;

  stage(0, 0);
  for (int k0 = 0; k0 < K; k0 += 32) {
    const int buf = (k0 >> 5) & 1;
    wait_async_all();
    __syncthreads();                 // tile[buf] ready; all waves done with buf^1
    if (k0 + 32 < K) stage(buf ^ 1, k0 + 32);
    if (k0 + 64 < K) {               // prefetch two tiles ahead into L2/WGP$
      int idx = threadIdx.x;
      int r = idx >> 2, c = (idx & 3) * 8;
      int gm = m0 + r; if (gm >= M) gm = M - 1;
      __builtin_prefetch(A + (size_t)gm * K + k0 + 64 + c, 0, 3);
      __builtin_prefetch(Bt + (size_t)(n0 + r) * K + k0 + 64 + c, 0, 3);
    }

    const h16* Ab = As + buf * (128 * 32);
    const h16* Bb = Bs + buf * (128 * 32);
    v16h bf[2];
#pragma unroll
    for (int nt = 0; nt < 2; ++nt)
      bf[nt] = ldfragB(Bb + (wn * 32 + nt * 16 + l15) * 32, hi);
#pragma unroll
    for (int mt = 0; mt < 4; ++mt) {
      v16h af = ldfragA(Ab + (wm * 64 + mt * 16 + l15) * 32, hi);
#pragma unroll
      for (int nt = 0; nt < 2; ++nt)
        acc[mt][nt] = wmma_f16(af, bf[nt], acc[mt][nt]);
    }
  }

#pragma unroll
  for (int mt = 0; mt < 4; ++mt)
#pragma unroll
    for (int nt = 0; nt < 2; ++nt)
#pragma unroll
      for (int r = 0; r < 8; ++r) {
        int gm = m0 + wm * 64 + mt * 16 + r + hi * 8;
        int gn = n0 + wn * 32 + nt * 16 + l15;
        if (gm < M) D[(size_t)gm * N + gn] = acc[mt][nt][r] + bias[gn];
      }
}

// ---------------------------------------------------------------------------
// Flash attention with decomposed relative-position bias.
// grid.x = S*NH sequences, grid.y = ceil(N/128). 8 waves x 16 query rows.
// Q,K: [seq][Npad][64] f16 ; Vt: [seq][64][Npad] f16 ; BH: [seq][N][H] ; BW: [seq][N][W].
// K/V tiles (32 keys) double-buffered in LDS with async fills, one barrier/step.
// out: f16 [4096][768], token mapping via window geometry (nwinW, wsz).
// ---------------------------------------------------------------------------
__global__ __launch_bounds__(256) void k_flash(const h16* __restrict__ Qg,
                                               const h16* __restrict__ Kg,
                                               const h16* __restrict__ Vg,
                                               const float* __restrict__ BHg,
                                               const float* __restrict__ BWg,
                                               h16* __restrict__ out,
                                               int N, int Npad, int Hn, int Wn,
                                               float scale, int nwinW, int wsz) {
  __shared__ __align__(16) h16 sK[2 * 32 * 64];   // [buf][key][d]
  __shared__ __align__(16) h16 sV[2 * 64 * 32];   // [buf][d][key]
  __shared__ __align__(16) h16 sP[8][16 * 32];

  const int seq = blockIdx.x;
  const int head = seq % 12;
  const int swin = seq / 12;
  const int wave = threadIdx.x >> 5;
  const int lane = threadIdx.x & 31;
  const int hi = lane >> 4;
  const int l15 = lane & 15;

  const h16* Qs = Qg + (size_t)seq * Npad * 64;
  const h16* Ks = Kg + (size_t)seq * Npad * 64;
  const h16* Vs = Vg + (size_t)seq * 64 * Npad;
  const float* BHs = BHg + (size_t)seq * N * Hn;
  const float* BWs = BWg + (size_t)seq * N * Wn;

  auto stage = [&](int buf, int k0) {
    int idx = threadIdx.x;
    const h16* gk = Ks + (size_t)k0 * 64 + idx * 8;
    h16* lk = sK + buf * 2048 + idx * 8;
    int d = idx >> 2, cc = (idx & 3) * 8;
    const h16* gv = Vs + (size_t)d * Npad + k0 + cc;
    h16* lv = sV + buf * 2048 + d * 32 + cc;
#if HAS_ASYNC
    async_cp16(gk, lk);
    async_cp16(gv, lv);
#else
    *(v8h*)lk = *(const v8h*)gk;
    *(v8h*)lv = *(const v8h*)gv;
#endif
  };

  const int qbase = blockIdx.y * 128 + wave * 16;
  int qrow = qbase + l15;
  if (qrow >= N) qrow = N - 1;                 // clamp (stores are guarded)
  const h16* qp = Qs + (size_t)qrow * 64;
  v16h qf0 = ldfragA(qp, hi);
  v16h qf1 = ldfragA(qp + 32, hi);

  float mrow[8], lrow[8];
  v8f O[4];
#pragma unroll
  for (int r = 0; r < 8; ++r) { mrow[r] = -INFINITY; lrow[r] = 0.f; }
#pragma unroll
  for (int j = 0; j < 4; ++j)
#pragma unroll
    for (int r = 0; r < 8; ++r) O[j][r] = 0.f;

  const int nkt = (N + 31) / 32;
  stage(0, 0);
  for (int kt = 0; kt < nkt; ++kt) {
    const int k0 = kt * 32;
    const int buf = kt & 1;
    wait_async_all();
    __syncthreads();                // tile[buf] ready; all waves done with buf^1
    if (kt + 1 < nkt) stage(buf ^ 1, k0 + 32);

    const h16* Kb = sK + buf * 2048;
    const h16* Vb = sV + buf * 2048;

    v8f Sv[2];
#pragma unroll
    for (int t = 0; t < 2; ++t) {
      const h16* krow = Kb + (t * 16 + l15) * 64;
      v16h kf0 = ldfragB(krow, hi);
      v16h kf1 = ldfragB(krow + 32, hi);
      v8f z;
#pragma unroll
      for (int e = 0; e < 8; ++e) z[e] = 0.f;
      z = wmma_f16(qf0, kf0, z);
      z = wmma_f16(qf1, kf1, z);
      Sv[t] = z;
    }

    float newm[8], sv[2][8];
#pragma unroll
    for (int r = 0; r < 8; ++r) newm[r] = mrow[r];
#pragma unroll
    for (int t = 0; t < 2; ++t)
#pragma unroll
      for (int r = 0; r < 8; ++r) {
        int key = k0 + t * 16 + l15;
        int row = qbase + r + hi * 8;
        float val;
        if (key < N && row < N) {
          int kh = key / Wn;
          int kw = key - kh * Wn;
          val = Sv[t][r] * scale + BHs[(size_t)row * Hn + kh] + BWs[(size_t)row * Wn + kw];
        } else {
          val = -1e30f;
        }
        sv[t][r] = val;
        newm[r] = fmaxf(newm[r], val);
      }
#pragma unroll
    for (int off = 1; off < 16; off <<= 1)
#pragma unroll
      for (int r = 0; r < 8; ++r)
        newm[r] = fmaxf(newm[r], __shfl_xor(newm[r], off, 32));

    float fr[8], rs[8];
#pragma unroll
    for (int r = 0; r < 8; ++r) {
      fr[r] = __expf(mrow[r] - newm[r]);
      mrow[r] = newm[r];
      rs[r] = 0.f;
    }
    h16* pw = &sP[wave][0];
#pragma unroll
    for (int t = 0; t < 2; ++t)
#pragma unroll
      for (int r = 0; r < 8; ++r) {
        float p = __expf(sv[t][r] - newm[r]);
        rs[r] += p;
        pw[(r + hi * 8) * 32 + t * 16 + l15] = (h16)p;   // transpose to A layout
      }
#pragma unroll
    for (int off = 1; off < 16; off <<= 1)
#pragma unroll
      for (int r = 0; r < 8; ++r) rs[r] += __shfl_xor(rs[r], off, 32);
#pragma unroll
    for (int r = 0; r < 8; ++r) lrow[r] = lrow[r] * fr[r] + rs[r];
#pragma unroll
    for (int j = 0; j < 4; ++j)
#pragma unroll
      for (int r = 0; r < 8; ++r) O[j][r] *= fr[r];

    v16h pf = ldfragA(&sP[wave][l15 * 32], hi);   // wave-private LDS, in-order
#pragma unroll
    for (int j = 0; j < 4; ++j) {
      v16h vf = ldfragB(Vb + (j * 16 + l15) * 32, hi);
      O[j] = wmma_f16(pf, vf, O[j]);
    }
  }

#pragma unroll
  for (int r = 0; r < 8; ++r) {
    int row = qbase + r + hi * 8;
    if (row < N) {
      float inv = 1.0f / lrow[r];
      int hq = row / Wn, wq = row - hq * Wn;
      int h = (swin / nwinW) * wsz + hq;
      int w = (swin % nwinW) * wsz + wq;
      if (h < 64 && w < 64) {
        h16* op = out + ((size_t)(h * 64 + w)) * 768 + head * 64 + l15;
#pragma unroll
        for (int j = 0; j < 4; ++j) op[j * 16] = (h16)(O[j][r] * inv);
      }
    }
  }
}

// ---------------------------------------------------------------------------
// Small support kernels
// ---------------------------------------------------------------------------
__global__ void k_im2col(const float* __restrict__ x, h16* __restrict__ A) {
  int id = blockIdx.x * 256 + threadIdx.x;
  if (id >= 4096 * 768) return;
  int k = id % 768, p = id / 768;
  int py = p >> 6, px = p & 63;
  int ky = k / 48, rem = k % 48, kx = rem / 3, c = rem % 3;
  A[id] = (h16)x[(size_t)c * 1048576 + (size_t)(py * 16 + ky) * 1024 + (px * 16 + kx)];
}

__global__ void k_convT(const float* __restrict__ W, h16* __restrict__ Wt, int K, int N) {
  long id = (long)blockIdx.x * 256 + threadIdx.x;
  if (id >= (long)K * N) return;
  int k = (int)(id % K);
  long n = id / K;
  Wt[id] = (h16)W[(size_t)k * N + n];    // Wt[n][k] = W[k][n]
}

__global__ void k_addpos(float* __restrict__ X, const float* __restrict__ D,
                         const float* __restrict__ pos, int n) {
  int id = blockIdx.x * 256 + threadIdx.x;
  if (id < n) X[id] = D[id] + pos[id];
}

__global__ void k_resid(float* __restrict__ X, const float* __restrict__ D, int n) {
  int id = blockIdx.x * 256 + threadIdx.x;
  if (id < n) X[id] += D[id];
}

__global__ void k_resid_out(const float* __restrict__ X, const float* __restrict__ D,
                            float* __restrict__ out, int n) {
  int id = blockIdx.x * 256 + threadIdx.x;
  if (id < n) out[id] = X[id] + D[id];
}

__global__ void k_gelu(const float* __restrict__ in, h16* __restrict__ out, int n) {
  int id = blockIdx.x * 256 + threadIdx.x;
  if (id < n) {
    float v = in[id];
    out[id] = (h16)(0.5f * v * (1.f + erff(v * 0.70710678118654752f)));
  }
}

__global__ __launch_bounds__(256) void k_ln(const float* __restrict__ X,
                                            const float* __restrict__ g,
                                            const float* __restrict__ b,
                                            h16* __restrict__ out) {
  __shared__ float rs[256], rs2[256];
  const int row = blockIdx.x;
  const int tid = threadIdx.x;
  const float* xr = X + (size_t)row * 768;
  float s = 0.f, s2 = 0.f;
  for (int c = tid; c < 768; c += 256) { float v = xr[c]; s += v; s2 += v * v; }
  rs[tid] = s; rs2[tid] = s2;
  __syncthreads();
  for (int off = 128; off; off >>= 1) {
    if (tid < off) { rs[tid] += rs[tid + off]; rs2[tid] += rs2[tid + off]; }
    __syncthreads();
  }
  float mean = rs[0] * (1.f / 768.f);
  float var = rs2[0] * (1.f / 768.f) - mean * mean;
  float inv = rsqrtf(var + 1e-5f);
  for (int c = tid; c < 768; c += 256)
    out[(size_t)row * 768 + c] = (h16)((xr[c] - mean) * inv * g[c] + b[c]);
}

__global__ void k_wingather(const h16* __restrict__ ln, h16* __restrict__ out) {
  int id = blockIdx.x * 256 + threadIdx.x;
  if (id >= 4900 * 96) return;
  int chunk = id % 96, wrow = id / 96;
  int win = wrow / 196, t = wrow % 196;
  int hq = t / 14, wq = t % 14;
  int h = (win / 5) * 14 + hq, w = (win % 5) * 14 + wq;
  v8h val;
  if (h < 64 && w < 64) {
    val = *(const v8h*)(ln + ((size_t)(h * 64 + w)) * 768 + chunk * 8);
  } else {
#pragma unroll
    for (int e = 0; e < 8; ++e) val[e] = (h16)0;
  }
  *(v8h*)(out + (size_t)wrow * 768 + chunk * 8) = val;
}

__global__ void k_scatter(const float* __restrict__ src, h16* __restrict__ Q,
                          h16* __restrict__ K, h16* __restrict__ V,
                          int S, int N, int Npad) {
  long id = (long)blockIdx.x * 256 + threadIdx.x;
  if (id >= (long)S * Npad * 2304) return;
  int j = (int)(id % 2304);
  long rt = id / 2304;
  int t = (int)(rt % Npad);
  int s = (int)(rt / Npad);
  int which = j / 768;
  int head = (j % 768) / 64;
  int d = j % 64;
  int seq = s * 12 + head;
  float v = (t < N) ? src[((size_t)s * N + t) * 2304 + j] : 0.f;
  h16 hv = (h16)v;
  if (which == 0)      Q[((size_t)seq * Npad + t) * 64 + d] = hv;
  else if (which == 1) K[((size_t)seq * Npad + t) * 64 + d] = hv;
  else                 V[((size_t)seq * 64 + d) * Npad + t] = hv;
}

__global__ void k_bias(const h16* __restrict__ Q, const float* __restrict__ relh,
                       const float* __restrict__ relw, float* __restrict__ BH,
                       float* __restrict__ BW, int SNH, int N, int Hn, int Wn,
                       int Npad) {
  long id = (long)blockIdx.x * 256 + threadIdx.x;
  int per = Hn + Wn;
  if (id >= (long)SNH * N * per) return;
  int j = (int)(id % per);
  long rn = id / per;
  int n = (int)(rn % N);
  int seq = (int)(rn / N);
  int hq = n / Wn, wq = n - hq * Wn;
  const h16* q = Q + ((size_t)seq * Npad + n) * 64;
  const float* tab;
  float* dst;
  if (j < Hn) {
    int idxr = hq - j + Hn - 1;
    tab = relh + (size_t)idxr * 64;
    dst = BH + ((size_t)seq * N + n) * Hn + j;
  } else {
    int kw = j - Hn;
    int idxr = wq - kw + Wn - 1;
    tab = relw + (size_t)idxr * 64;
    dst = BW + ((size_t)seq * N + n) * Wn + kw;
  }
  float acc = 0.f;
#pragma unroll
  for (int d = 0; d < 64; ++d) acc += (float)q[d] * tab[d];
  *dst = acc;
}

// ---------------------------------------------------------------------------
// Workspace layout (bytes, 256-aligned)
// ---------------------------------------------------------------------------
static constexpr size_t OFF_X    = 0;                              // 4096*768 f32
static constexpr size_t OFF_BIG  = OFF_X    + 12582912;            // <=4096*3072 f32
static constexpr size_t OFF_ACT  = OFF_BIG  + 50331648;            // <=4096*3072 f16
static constexpr size_t OFF_ACTW = OFF_ACT  + 25165824;            // 4900*768 f16
static constexpr size_t OFF_Q    = OFF_ACTW + 7526400;             // 67200*64 f16
static constexpr size_t OFF_K    = OFF_Q    + 8601600;
static constexpr size_t OFF_VT   = OFF_K    + 8601600;
static constexpr size_t OFF_BH   = OFF_VT   + 8601600;             // 12*4096*64 f32
static constexpr size_t OFF_BW   = OFF_BH   + 12582912;
static constexpr size_t OFF_W16  = OFF_BW   + 12582912;            // 768*3072 f16

extern "C" void kernel_launch(void* const* d_in, const int* in_sizes, int n_in,
                              void* d_out, int out_size, void* d_ws, size_t ws_size,
                              hipStream_t stream) {
  const float* x       = (const float*)d_in[0];
  const float* patch_w = (const float*)d_in[1];
  const float* patch_b = (const float*)d_in[2];
  const float* pos     = (const float*)d_in[3];
  const float* P1[14];
  const float* P2[14];
  for (int i = 0; i < 14; ++i) {
    P1[i] = (const float*)d_in[4 + i];
    P2[i] = (const float*)d_in[18 + i];
  }
  // per-block params: 0 ln1g 1 ln1b 2 qkvw 3 qkvb 4 projw 5 projb 6 relh 7 relw
  //                   8 ln2g 9 ln2b 10 mw1 11 mb1 12 mw2 13 mb2

  char* ws = (char*)d_ws;
  float* X   = (float*)(ws + OFF_X);
  float* BIG = (float*)(ws + OFF_BIG);
  h16* ACT   = (h16*)(ws + OFF_ACT);
  h16* ACTW  = (h16*)(ws + OFF_ACTW);
  h16* Qb    = (h16*)(ws + OFF_Q);
  h16* Kb    = (h16*)(ws + OFF_K);
  h16* Vb    = (h16*)(ws + OFF_VT);
  float* BH  = (float*)(ws + OFF_BH);
  float* BW  = (float*)(ws + OFF_BW);
  h16* W16   = (h16*)(ws + OFF_W16);

  auto cdiv = [](long a, long b) { return (unsigned)((a + b - 1) / b); };
  const int NT = 4096 * 768;  // tokens*channels

  // ---- patch embed: im2col + GEMM + bias + pos_embed ----
  k_im2col<<<cdiv(4096L * 768, 256), 256, 0, stream>>>(x, ACTW);
  k_convT<<<cdiv(768L * 768, 256), 256, 0, stream>>>(patch_w, W16, 768, 768);
  k_gemm<<<dim3(6, 32), 256, 0, stream>>>(ACTW, W16, patch_b, BIG, 4096, 768, 768);
  k_addpos<<<cdiv(NT, 256), 256, 0, stream>>>(X, BIG, pos, NT);

  for (int blk = 0; blk < 2; ++blk) {
    const float** P = blk ? P2 : P1;
    const bool win = (blk == 0);
    const int S = win ? 25 : 1;
    const int N = win ? 196 : 4096;
    const int Npad = win ? 224 : 4096;
    const int Hn = win ? 14 : 64;
    const int R = S * N;

    // LN1 -> f16; optional window partition (zero-padded to 70x70, 25 windows)
    k_ln<<<4096, 256, 0, stream>>>(X, P[0], P[1], ACT);
    const h16* qa = ACT;
    if (win) {
      k_wingather<<<cdiv(4900L * 96, 256), 256, 0, stream>>>(ACT, ACTW);
      qa = ACTW;
    }

    // QKV GEMM + scatter to Q/K/V^T (f16), pads zeroed
    k_convT<<<cdiv(768L * 2304, 256), 256, 0, stream>>>(P[2], W16, 768, 2304);
    k_gemm<<<dim3(18, cdiv(R, 128)), 256, 0, stream>>>(qa, W16, P[3], BIG, R, 2304, 768);
    k_scatter<<<cdiv((long)S * Npad * 2304, 256), 256, 0, stream>>>(BIG, Qb, Kb, Vb,
                                                                    S, N, Npad);

    // decomposed relative position bias tables, then flash attention
    k_bias<<<cdiv((long)S * 12 * N * (2 * Hn), 256), 256, 0, stream>>>(
        Qb, P[6], P[7], BH, BW, S * 12, N, Hn, Hn, Npad);
    k_flash<<<dim3(S * 12, cdiv(N, 128)), 256, 0, stream>>>(
        Qb, Kb, Vb, BH, BW, ACT, N, Npad, Hn, Hn, 0.125f, win ? 5 : 1,
        win ? 14 : 64);

    // proj + residual
    k_convT<<<cdiv(768L * 768, 256), 256, 0, stream>>>(P[4], W16, 768, 768);
    k_gemm<<<dim3(6, 32), 256, 0, stream>>>(ACT, W16, P[5], BIG, 4096, 768, 768);
    k_resid<<<cdiv(NT, 256), 256, 0, stream>>>(X, BIG, NT);

    // MLP: LN2 -> fc1 -> gelu -> fc2 -> residual (last one writes d_out)
    k_ln<<<4096, 256, 0, stream>>>(X, P[8], P[9], ACT);
    k_convT<<<cdiv(768L * 3072, 256), 256, 0, stream>>>(P[10], W16, 768, 3072);
    k_gemm<<<dim3(24, 32), 256, 0, stream>>>(ACT, W16, P[11], BIG, 4096, 3072, 768);
    k_gelu<<<cdiv(4096L * 3072, 256), 256, 0, stream>>>(BIG, ACT, 4096 * 3072);
    k_convT<<<cdiv(3072L * 768, 256), 256, 0, stream>>>(P[12], W16, 3072, 768);
    k_gemm<<<dim3(6, 32), 256, 0, stream>>>(ACT, W16, P[13], BIG, 4096, 768, 3072);
    if (blk == 0)
      k_resid<<<cdiv(NT, 256), 256, 0, stream>>>(X, BIG, NT);
    else
      k_resid_out<<<cdiv(NT, 256), 256, 0, stream>>>(X, BIG, (float*)d_out, NT);
  }
}
